// GraphNet_56349970923934
// MI455X (gfx1250) — compile-verified, matrix operand order checked by reading, not run on hardware
//
#include <hip/hip_runtime.h>
#include <math.h>

typedef __bf16 bf16;
typedef __attribute__((ext_vector_type(16))) __bf16 v16bf;
typedef __attribute__((ext_vector_type(8)))  float  v8f;

constexpr int kH = 768;     // hidden
constexpr int kB = 512;     // batch
constexpr int kP = 32;      // parents (scan length)
constexpr int kN = 64;      // nodes per batch
constexpr int kT = 128;     // tokens per batch
constexpr int kG = 3 * kH;  // 2304 (GRU gate width)

// ---------------------------------------------------------------------------
// WMMA fragment loader: 16-bit A-layout (ISA 7.12.2).
// lane (0..31): row = lane&15, hi = lane>>4.
// v16bf elems e: vgpr v=e>>1, pair p=e&1 -> K = (v>=4?16:0) + hi*8 + (v&3)*2 + p
// => two contiguous 16B loads at k + hi*8 and k + 16 + hi*8.
// Same pattern serves the B operand when W is stored [Nout, K] row-major
// (lane holds column n = lane&15 of B, i.e. row n of W).
// ---------------------------------------------------------------------------
__device__ __forceinline__ v16bf load_frag(const bf16* __restrict__ rowk, int hi) {
  union { v16bf v; uint4 q[2]; } u;
  u.q[0] = *(const uint4*)(rowk + hi * 8);
  u.q[1] = *(const uint4*)(rowk + 16 + hi * 8);
  return u.v;
}

// Four WMMAs sharing one A fragment; reuse_a hint set on trailing three
// (previous instruction is the identical WMMA shape with the same A regs).
__device__ __forceinline__ void wmma4(const v16bf& fa, const v16bf fb[4], v8f acc[4]) {
  acc[0] = __builtin_amdgcn_wmma_f32_16x16x32_bf16(
      false, fa, false, fb[0], (short)0, acc[0], false, false);
  acc[1] = __builtin_amdgcn_wmma_f32_16x16x32_bf16(
      false, fa, false, fb[1], (short)0, acc[1], true, false);
  acc[2] = __builtin_amdgcn_wmma_f32_16x16x32_bf16(
      false, fa, false, fb[2], (short)0, acc[2], true, false);
  acc[3] = __builtin_amdgcn_wmma_f32_16x16x32_bf16(
      false, fa, false, fb[3], (short)0, acc[3], true, false);
}

// ---------------------------------------------------------------------------
// Copy-free, fully unrolled, software-pipelined K loop (K compile-time).
// Slice s+1 fragments are loaded between the two WMMA groups; full unroll
// makes every buffer assignment a fresh SSA value (no v_mov rotation) and
// every load a base + immediate-offset access.
// ---------------------------------------------------------------------------
template <int K>
__device__ __forceinline__ void gemm_core(
    const bf16* __restrict__ arow,
    const bf16* __restrict__ w0, const bf16* __restrict__ w1,
    const bf16* __restrict__ w2, const bf16* __restrict__ w3,
    int hi, v8f acc[4])
{
  constexpr int S = K / 32;           // #slices (even for all instantiations)
  static_assert((S & 1) == 0, "K/32 must be even");

  v16bf a0 = load_frag(arow, hi);
  v16bf b0[4] = {load_frag(w0, hi), load_frag(w1, hi),
                 load_frag(w2, hi), load_frag(w3, hi)};

#pragma unroll
  for (int s = 1; s + 1 < S; s += 2) {
    v16bf a1 = load_frag(arow + s * 32, hi);
    v16bf b1[4] = {load_frag(w0 + s * 32, hi), load_frag(w1 + s * 32, hi),
                   load_frag(w2 + s * 32, hi), load_frag(w3 + s * 32, hi)};
    wmma4(a0, b0, acc);

    a0 = load_frag(arow + (s + 1) * 32, hi);
    b0[0] = load_frag(w0 + (s + 1) * 32, hi);
    b0[1] = load_frag(w1 + (s + 1) * 32, hi);
    b0[2] = load_frag(w2 + (s + 1) * 32, hi);
    b0[3] = load_frag(w3 + (s + 1) * 32, hi);
    wmma4(a1, b1, acc);
  }

  // epilogue: last slice
  {
    constexpr int s = S - 1;
    v16bf a1 = load_frag(arow + s * 32, hi);
    v16bf b1[4] = {load_frag(w0 + s * 32, hi), load_frag(w1 + s * 32, hi),
                   load_frag(w2 + s * 32, hi), load_frag(w3 + s * 32, hi)};
    wmma4(a0, b0, acc);
    wmma4(a1, b1, acc);
  }
}

// ---------------------------------------------------------------------------
// Generic GEMM: out[M,Nout] = A[M,K] * W[Nout,K]^T (+bias), f32 out,
// optional bf16 mirror at obf[row*ldob + col_off + col].
// Block tile: 128(M) x 64(N), 256 threads = 8 waves; each wave: 16x64 via
// 4 x v_wmma_f32_16x16x32_bf16 per K=32 slice.
// ---------------------------------------------------------------------------
template <int K>
__global__ __launch_bounds__(256) void gemm_bf16_kernel(
    const bf16* __restrict__ A, int lda,
    const bf16* __restrict__ W, int ldw,
    const float* __restrict__ bias,
    float* __restrict__ out, int ldo,
    bf16* __restrict__ obf, int ldob, int col_off)
{
  const int lane = threadIdx.x & 31;
  const int wave = threadIdx.x >> 5;
  const int hi = lane >> 4;
  const int lo = lane & 15;
  const int tileN = blockIdx.x * 64;
  const int tileM = blockIdx.y * 128;

  const bf16* arow = A + (size_t)(tileM + wave * 16 + lo) * lda;
  const bf16* w0 = W + (size_t)(tileN + 0 * 16 + lo) * ldw;
  const bf16* w1 = W + (size_t)(tileN + 1 * 16 + lo) * ldw;
  const bf16* w2 = W + (size_t)(tileN + 2 * 16 + lo) * ldw;
  const bf16* w3 = W + (size_t)(tileN + 3 * 16 + lo) * ldw;

  v8f acc[4] = {v8f{}, v8f{}, v8f{}, v8f{}};
  gemm_core<K>(arow, w0, w1, w2, w3, hi, acc);

  const int row0 = tileM + wave * 16 + hi * 8;   // D layout: M = vgpr + 8*hi
#pragma unroll
  for (int f = 0; f < 4; ++f) {
    const int col = tileN + f * 16 + lo;         // D layout: N = lane&15
    const float bv = bias ? bias[col] : 0.f;
#pragma unroll
    for (int r = 0; r < 8; ++r) {
      float v = acc[f][r] + bv;
      out[(size_t)(row0 + r) * ldo + col] = v;
      if (obf) obf[(size_t)(row0 + r) * ldob + col_off + col] = (bf16)v;
    }
  }
}

// ---------------------------------------------------------------------------
// Fused scan-step GEMM: virtual A of 1024 rows; rows [0,512) = h (uses Wh),
// rows [512,1024) = parents[:,t,:] (uses Wi). Out = gbuf[1024, 2304].
// ---------------------------------------------------------------------------
__global__ __launch_bounds__(256) void gemm_step_kernel(
    const bf16* __restrict__ Hbf,     // [512, 768]
    const bf16* __restrict__ Pbf,     // [512, 32, 768]
    int t,
    const bf16* __restrict__ Whb,     // [2304, 768]
    const bf16* __restrict__ Wib,     // [2304, 768]
    float* __restrict__ gout)         // [1024, 2304]
{
  const int lane = threadIdx.x & 31;
  const int wave = threadIdx.x >> 5;
  const int hi = lane >> 4;
  const int lo = lane & 15;
  const int tileN = blockIdx.x * 64;
  const int tileM = blockIdx.y * 128;

  const int m = tileM + wave * 16 + lo;
  const bf16* arow = (m < kB)
      ? Hbf + (size_t)m * kH
      : Pbf + ((size_t)(m - kB) * kP + t) * kH;
  const bf16* W = (tileM < kB) ? Whb : Wib;   // uniform per block
  const bf16* w0 = W + (size_t)(tileN + 0 * 16 + lo) * kH;
  const bf16* w1 = W + (size_t)(tileN + 1 * 16 + lo) * kH;
  const bf16* w2 = W + (size_t)(tileN + 2 * 16 + lo) * kH;
  const bf16* w3 = W + (size_t)(tileN + 3 * 16 + lo) * kH;

  v8f acc[4] = {v8f{}, v8f{}, v8f{}, v8f{}};
  gemm_core<kH>(arow, w0, w1, w2, w3, hi, acc);

  const int row0 = tileM + wave * 16 + hi * 8;
#pragma unroll
  for (int f = 0; f < 4; ++f) {
    const int col = tileN + f * 16 + lo;
#pragma unroll
    for (int r = 0; r < 8; ++r)
      gout[(size_t)(row0 + r) * kG + col] = acc[f][r];
  }
}

// ---------------------------------------------------------------------------
// f32 -> bf16 convert (weights)
// ---------------------------------------------------------------------------
__global__ void f32_to_bf16_kernel(const float* __restrict__ src,
                                   bf16* __restrict__ dst, int n) {
  for (int i = blockIdx.x * 256 + threadIdx.x; i < n; i += gridDim.x * 256)
    dst[i] = (bf16)src[i];
}

// ---------------------------------------------------------------------------
// Gather child + parents, init h = 0. One block per batch element.
// ---------------------------------------------------------------------------
__global__ __launch_bounds__(256) void gather_init_kernel(
    const float* __restrict__ nodes, const int* __restrict__ his,
    const int* __restrict__ child_id,
    bf16* __restrict__ parents_bf, float* __restrict__ child_f,
    float* __restrict__ h, bf16* __restrict__ h_bf)
{
  const int b = blockIdx.x;
  const float* nb = nodes + (size_t)b * kN * kH;
  const int cid = child_id[b];
  for (int j = threadIdx.x; j < kH; j += 256) {
    child_f[(size_t)b * kH + j] = nb[(size_t)cid * kH + j];
    h[(size_t)b * kH + j] = 0.f;
    h_bf[(size_t)b * kH + j] = (bf16)0.f;
  }
  for (int idx = threadIdx.x; idx < kP * kH; idx += 256) {
    const int p = idx / kH, j = idx - p * kH;
    const int src = his[b * kP + p];
    parents_bf[((size_t)b * kP + p) * kH + j] = (bf16)nb[(size_t)src * kH + j];
  }
}

// ---------------------------------------------------------------------------
// Elementwise GRU gate update for step t (consumes gbuf from gemm_step).
// ---------------------------------------------------------------------------
__global__ __launch_bounds__(256) void gru_update_kernel(
    const float* __restrict__ gbuf, const float* __restrict__ bi,
    const float* __restrict__ bh, float* __restrict__ h,
    bf16* __restrict__ h_bf, float* __restrict__ outputs,
    const int* __restrict__ parents_num, int t)
{
  const int b = blockIdx.x;
  const bool act = t < parents_num[b];
  const float* gh = gbuf + (size_t)b * kG;
  const float* gi = gbuf + (size_t)(kB + b) * kG;
  for (int j = threadIdx.x; j < kH; j += 256) {
    const float hr = gh[j] + bh[j];
    const float hz = gh[kH + j] + bh[kH + j];
    const float hn = gh[2 * kH + j] + bh[2 * kH + j];
    const float ir = gi[j] + bi[j];
    const float iz = gi[kH + j] + bi[kH + j];
    const float in_ = gi[2 * kH + j] + bi[2 * kH + j];
    const float r = 1.f / (1.f + __expf(-(ir + hr)));
    const float z = 1.f / (1.f + __expf(-(iz + hz)));
    const float n = tanhf(in_ + r * hn);
    const float hp = h[(size_t)b * kH + j];
    const float hnew = (1.f - z) * n + z * hp;
    const float hout = act ? hnew : hp;
    h[(size_t)b * kH + j] = hout;
    h_bf[(size_t)b * kH + j] = (bf16)hout;
    outputs[((size_t)b * kP + t) * kH + j] = hout;
  }
}

// ---------------------------------------------------------------------------
// Masked attention over GRU outputs -> bf16 att (input of wqc GEMM).
// ---------------------------------------------------------------------------
__global__ __launch_bounds__(256) void parent_attn_kernel(
    const float* __restrict__ outputs, const float* __restrict__ child_f,
    const int* __restrict__ parents_num, bf16* __restrict__ att_bf)
{
  __shared__ float sc[kP];
  __shared__ float prob[kP];
  const int b = blockIdx.x;
  const int tid = threadIdx.x;
  const int p = tid >> 3, sub = tid & 7;     // 32 scores x 8 partial lanes
  const float* orow = outputs + ((size_t)b * kP + p) * kH;
  const float* ch = child_f + (size_t)b * kH;
  float s = 0.f;
  for (int i = 0; i < kH / 8; ++i) {
    const int j = sub * (kH / 8) + i;
    s += orow[j] * ch[j];
  }
  s += __shfl_down(s, 4, 8);
  s += __shfl_down(s, 2, 8);
  s += __shfl_down(s, 1, 8);
  if (sub == 0) sc[p] = (p < parents_num[b]) ? s : -INFINITY;
  __syncthreads();
  if (tid < 32) {
    float v = sc[tid];
    float m = v;
    for (int o = 16; o; o >>= 1) m = fmaxf(m, __shfl_xor(m, o, 32));
    float e = __expf(v - m);
    float sum = e;
    for (int o = 16; o; o >>= 1) sum += __shfl_xor(sum, o, 32);
    prob[tid] = e / sum;
  }
  __syncthreads();
  for (int j = tid; j < kH; j += 256) {
    float a = 0.f;
#pragma unroll 8
    for (int q = 0; q < kP; ++q)
      a += prob[q] * outputs[((size_t)b * kP + q) * kH + j];
    att_bf[(size_t)b * kH + j] = (bf16)a;
  }
}

// ---------------------------------------------------------------------------
// Softmax attention over tokens -> que_emb, written as bf16 into the low half
// of the concat buffer ctxin[b, 0:768].
// ---------------------------------------------------------------------------
__global__ __launch_bounds__(256) void token_attn_kernel(
    const float* __restrict__ tokens, const float* __restrict__ child_f,
    bf16* __restrict__ ctxin)
{
  __shared__ float sc[kT];
  const int b = blockIdx.x;
  const int tid = threadIdx.x;
  const int t = tid >> 1, sub = tid & 1;     // 128 scores x 2 partial lanes
  const float* trow = tokens + ((size_t)b * kT + t) * kH;
  const float* ch = child_f + (size_t)b * kH;
  float s = 0.f;
  for (int i = 0; i < kH / 2; ++i) {
    const int j = sub * (kH / 2) + i;
    s += trow[j] * ch[j];
  }
  s += __shfl_down(s, 1, 2);
  if (sub == 0) sc[t] = s;
  __syncthreads();
  if (tid < 32) {
    float v0 = sc[tid], v1 = sc[tid + 32], v2 = sc[tid + 64], v3 = sc[tid + 96];
    float m = fmaxf(fmaxf(v0, v1), fmaxf(v2, v3));
    for (int o = 16; o; o >>= 1) m = fmaxf(m, __shfl_xor(m, o, 32));
    float e0 = __expf(v0 - m), e1 = __expf(v1 - m);
    float e2 = __expf(v2 - m), e3 = __expf(v3 - m);
    float sum = e0 + e1 + e2 + e3;
    for (int o = 16; o; o >>= 1) sum += __shfl_xor(sum, o, 32);
    const float inv = 1.f / sum;
    sc[tid] = e0 * inv; sc[tid + 32] = e1 * inv;
    sc[tid + 64] = e2 * inv; sc[tid + 96] = e3 * inv;
  }
  __syncthreads();
  for (int j = tid; j < kH; j += 256) {
    float a = 0.f;
    for (int tt = 0; tt < kT; ++tt)
      a += sc[tt] * tokens[((size_t)b * kT + tt) * kH + j];
    ctxin[(size_t)b * (2 * kH) + j] = (bf16)a;   // que half of concat
  }
}

// ---------------------------------------------------------------------------
// pred[b] = sigmoid(dot(ctx_out[b], child[b])). One wave per batch row.
// ---------------------------------------------------------------------------
__global__ __launch_bounds__(256) void final_pred_kernel(
    const float* __restrict__ ctx_out, const float* __restrict__ child_f,
    float* __restrict__ pred)
{
  const int b = blockIdx.x * 8 + (threadIdx.x >> 5);
  const int lane = threadIdx.x & 31;
  const float* c = ctx_out + (size_t)b * kH;
  const float* ch = child_f + (size_t)b * kH;
  float s = 0.f;
  for (int j = lane; j < kH; j += 32) s += c[j] * ch[j];
  for (int o = 16; o; o >>= 1) s += __shfl_xor(s, o, 32);
  if (lane == 0) pred[b] = 1.f / (1.f + __expf(-s));
}

// ---------------------------------------------------------------------------
extern "C" void kernel_launch(void* const* d_in, const int* in_sizes, int n_in,
                              void* d_out, int out_size, void* d_ws, size_t ws_size,
                              hipStream_t stream)
{
  (void)in_sizes; (void)n_in; (void)out_size;
  const float* tokens      = (const float*)d_in[0];
  const float* nodes       = (const float*)d_in[1];
  const int*   his         = (const int*)d_in[2];
  const int*   parents_num = (const int*)d_in[3];
  const int*   child_id    = (const int*)d_in[4];
  const float* gru_Wi      = (const float*)d_in[5];
  const float* gru_Wh      = (const float*)d_in[6];
  const float* gru_bi      = (const float*)d_in[7];
  const float* gru_bh      = (const float*)d_in[8];
  const float* wqc_W       = (const float*)d_in[9];
  const float* wqc_b       = (const float*)d_in[10];
  const float* ctx_W       = (const float*)d_in[11];
  const float* ctx_b       = (const float*)d_in[12];
  float* pred = (float*)d_out;

  char* ws = (char*)d_ws;
  size_t off = 0;
  auto alloc = [&](size_t bytes) -> char* {
    char* p = ws + off;
    off = (off + bytes + 255) & ~(size_t)255;
    return p;
  };
  bf16*  Wi_bf      = (bf16*)alloc((size_t)kG * kH * 2);
  bf16*  Wh_bf      = (bf16*)alloc((size_t)kG * kH * 2);
  bf16*  wqc_bf     = (bf16*)alloc((size_t)kH * kH * 2);
  bf16*  ctx_bf     = (bf16*)alloc((size_t)kH * 2 * kH * 2);
  bf16*  parents_bf = (bf16*)alloc((size_t)kB * kP * kH * 2);
  float* child_f    = (float*)alloc((size_t)kB * kH * 4);
  float* h_f        = (float*)alloc((size_t)kB * kH * 4);
  bf16*  h_bf       = (bf16*)alloc((size_t)kB * kH * 2);
  float* gbuf       = (float*)alloc((size_t)2 * kB * kG * 4);
  float* outputs    = (float*)alloc((size_t)kB * kP * kH * 4);
  bf16*  att_bf     = (bf16*)alloc((size_t)kB * kH * 2);
  float* pemb_f     = (float*)alloc((size_t)kB * kH * 4);
  bf16*  ctxin      = (bf16*)alloc((size_t)kB * 2 * kH * 2);
  float* ctxout     = (float*)alloc((size_t)kB * kH * 4);
  if (off > ws_size) return;   // workspace too small; avoid OOB writes

  // 1) weight conversion f32 -> bf16
  const int nWi = kG * kH, nWqc = kH * kH, nCtx = kH * 2 * kH;
  f32_to_bf16_kernel<<<(nWi + 255) / 256, 256, 0, stream>>>(gru_Wi, Wi_bf, nWi);
  f32_to_bf16_kernel<<<(nWi + 255) / 256, 256, 0, stream>>>(gru_Wh, Wh_bf, nWi);
  f32_to_bf16_kernel<<<(nWqc + 255) / 256, 256, 0, stream>>>(wqc_W, wqc_bf, nWqc);
  f32_to_bf16_kernel<<<(nCtx + 255) / 256, 256, 0, stream>>>(ctx_W, ctx_bf, nCtx);

  // 2) gather parents/child, h = 0
  gather_init_kernel<<<kB, 256, 0, stream>>>(nodes, his, child_id,
                                             parents_bf, child_f, h_f, h_bf);

  // 3) token attention (independent of the scan; bandwidth-dominant)
  token_attn_kernel<<<kB, 256, 0, stream>>>(tokens, child_f, ctxin);

  // 4) masked GRU scan: per step one fused WMMA GEMM (gi_t & gh_t) + gates
  for (int t = 0; t < kP; ++t) {
    gemm_step_kernel<<<dim3(kG / 64, (2 * kB) / 128), 256, 0, stream>>>(
        h_bf, parents_bf, t, Wh_bf, Wi_bf, gbuf);
    gru_update_kernel<<<kB, 256, 0, stream>>>(
        gbuf, gru_bi, gru_bh, h_f, h_bf, outputs, parents_num, t);
  }

  // 5) masked attention over GRU outputs
  parent_attn_kernel<<<kB, 256, 0, stream>>>(outputs, child_f, parents_num, att_bf);

  // 6) parents_emb = att @ wqc_W^T + b  (bf16 mirror -> high half of concat)
  gemm_bf16_kernel<kH><<<dim3(kH / 64, kB / 128), 256, 0, stream>>>(
      att_bf, kH, wqc_bf, kH, wqc_b, pemb_f, kH, ctxin, 2 * kH, kH);

  // 7) context = [que, parents] @ ctx_W^T + b
  gemm_bf16_kernel<2 * kH><<<dim3(kH / 64, kB / 128), 256, 0, stream>>>(
      ctxin, 2 * kH, ctx_bf, 2 * kH, ctx_b, ctxout, kH, (bf16*)nullptr, 0, 0);

  // 8) pred = sigmoid(<context, child>)
  final_pred_kernel<<<kB / 8, 256, 0, stream>>>(ctxout, child_f, pred);
}
